// LightweightAttention_78967268704399
// MI455X (gfx1250) — matrix operand order repaired and checked by reading, not compiled, not run on hardware
//
#include <hip/hip_runtime.h>
#include <math.h>

#define DIMV 1024
#define HEADS 16
#define HD 64
#define BATCH 4
#define SEQL 4096
#define M_TOT (BATCH * SEQL)  // 16384

typedef __attribute__((ext_vector_type(16))) __bf16 v16bf;
typedef __attribute__((ext_vector_type(8)))  float  v8f;

union FragU { uint4 q[2]; v16bf v; };
union Pack4 { uint2 u; __bf16 h[4]; };

__device__ __forceinline__ v8f zero8() {
  v8f z;
#pragma unroll
  for (int i = 0; i < 8; ++i) z[i] = 0.0f;
  return z;
}

__device__ __forceinline__ v8f bwmma(v16bf a, v16bf b, v8f c) {
  // D = A(16x32 bf16) * B(32x16 bf16) + C(16x16 f32)
  return __builtin_amdgcn_wmma_f32_16x16x32_bf16(false, a, false, b, (short)0, c,
                                                 false, false);
}

// A fragment: LDS row m holds K contiguous. Per ISA layout:
// lane<16: elems0-7=K0..7, elems8-15=K16..23 ; lane>=16: K8..15 / K24..31
__device__ __forceinline__ v16bf frag_a(const __bf16* lds, int pitch, int mbase,
                                        int kbase, int lane) {
  int m = mbase + (lane & 15);
  int k0 = (lane >= 16) ? 8 : 0;
  const char* p = (const char*)(lds + m * pitch + kbase + k0);
  FragU f;
  f.q[0] = *(const uint4*)(p);
  f.q[1] = *(const uint4*)(p + 32);  // +16 elements
  return f.v;
}

// B fragment: LDS stores Bt (row n holds K contiguous).
// lane<16: elems0-7=K0..7, elems8-15=K16..23 of column n ; lane>=16: K8..15 / K24..31
__device__ __forceinline__ v16bf frag_b(const __bf16* lds, int pitch, int nbase,
                                        int kbase, int lane) {
  int n = nbase + (lane & 15);
  int kb = (lane >= 16) ? 8 : 0;
  const char* p = (const char*)(lds + n * pitch + kbase + kb);
  FragU f;
  f.q[0] = *(const uint4*)(p);
  f.q[1] = *(const uint4*)(p + 32);  // +16 elements
  return f.v;
}

// vectorized 4-element loads (fp32 or bf16 source) into fp32 regs
__device__ __forceinline__ void load4(const float* p, float o[4]) {
  float4 v = *(const float4*)p;
  o[0] = v.x; o[1] = v.y; o[2] = v.z; o[3] = v.w;
}
__device__ __forceinline__ void load4(const __bf16* p, float o[4]) {
  Pack4 q;
  q.u = *(const uint2*)p;
#pragma unroll
  for (int i = 0; i < 4; ++i) o[i] = (float)q.h[i];
}
// packed 4x bf16 (8-byte) LDS store
__device__ __forceinline__ void store4bf(__bf16* p, const float v[4]) {
  Pack4 q;
#pragma unroll
  for (int i = 0; i < 4; ++i) q.h[i] = (__bf16)v[i];
  *(uint2*)p = q.u;
}

// ---------------------------------------------------------------------------
// C[M,N] = A[M,K] * W[N,K]^T + bias   (bf16 WMMA, fp32 accumulate)
// Block tile 256x128, 256 thr = 8 waves (4 row x 2 col groups), each wave a
// 64x64 tile (4x4 accum tiles, 16 WMMA / 32-k step). Double-buffered LDS with
// register-staged prefetch of the next k-tile.
// ---------------------------------------------------------------------------
template <typename AT>
__global__ __launch_bounds__(256) void gemm_xwt_bias(
    const AT* __restrict__ A, const float* __restrict__ W,
    const float* __restrict__ bias, float* __restrict__ C, int M, int N, int K) {
  __shared__ __align__(16) __bf16 sA[2][256 * 40];
  __shared__ __align__(16) __bf16 sB[2][128 * 40];
  const int t = threadIdx.x;
  const int lane = t & 31;
  const int w = t >> 5;
  const int wr = w & 3;   // 64-row group
  const int wc = w >> 2;  // 64-col group
  const int n0 = blockIdx.x * 128;
  const int m0 = blockIdx.y * 256;
  // staging: 8 threads span 32 cols (float4), 32 row-bases
  const int sc4 = (t & 7) * 4;
  const int sr = t >> 3;  // 0..31

  float ra[8][4], rb[4][4];
  auto fetch = [&](int k0) {
#pragma unroll
    for (int j = 0; j < 8; ++j)
      load4(A + (size_t)(m0 + sr + j * 32) * K + k0 + sc4, ra[j]);
#pragma unroll
    for (int j = 0; j < 4; ++j)
      load4(W + (size_t)(n0 + sr + j * 32) * K + k0 + sc4, rb[j]);
  };
  auto commit = [&](int buf) {
#pragma unroll
    for (int j = 0; j < 8; ++j)
      store4bf(&sA[buf][(sr + j * 32) * 40 + sc4], ra[j]);
#pragma unroll
    for (int j = 0; j < 4; ++j)
      store4bf(&sB[buf][(sr + j * 32) * 40 + sc4], rb[j]);
  };

  v8f acc[4][4];
#pragma unroll
  for (int i = 0; i < 4; ++i)
#pragma unroll
    for (int j = 0; j < 4; ++j) acc[i][j] = zero8();

  fetch(0);
  commit(0);
  __syncthreads();

  const int nsteps = K >> 5;
  for (int s = 0; s < nsteps; ++s) {
    const int cur = s & 1;
    if (s + 1 < nsteps) fetch((s + 1) * 32);  // overlap next tile w/ compute
    if (s + 2 < nsteps) {                     // warm L2 for tile after that
      __builtin_prefetch(A + (size_t)(m0 + sr) * K + (s + 2) * 32 + sc4, 0, 1);
      __builtin_prefetch(W + (size_t)(n0 + sr) * K + (s + 2) * 32 + sc4, 0, 1);
    }
    // hoist all B fragments, then burst 16 WMMAs
    v16bf bfr[4];
#pragma unroll
    for (int nt = 0; nt < 4; ++nt)
      bfr[nt] = frag_b(sB[cur], 40, wc * 64 + nt * 16, 0, lane);
#pragma unroll
    for (int mt = 0; mt < 4; ++mt) {
      v16bf a = frag_a(sA[cur], 40, wr * 64 + mt * 16, 0, lane);
#pragma unroll
      for (int nt = 0; nt < 4; ++nt)
        acc[mt][nt] = bwmma(a, bfr[nt], acc[mt][nt]);
    }
    if (s + 1 < nsteps) commit(cur ^ 1);
    __syncthreads();
  }

  const int hi8 = (lane >= 16) ? 8 : 0;
  const int nl = lane & 15;
#pragma unroll
  for (int mt = 0; mt < 4; ++mt)
#pragma unroll
    for (int nt = 0; nt < 4; ++nt) {
      int col = n0 + wc * 64 + nt * 16 + nl;
      float bv = bias[col];
#pragma unroll
      for (int r = 0; r < 8; ++r) {
        int row = m0 + wr * 64 + mt * 16 + hi8 + r;
        C[(size_t)row * N + col] = acc[mt][nt][r] + bv;
      }
    }
}

// ---------------------------------------------------------------------------
// Performer feature map: xp = exp((x/8) @ R - 0.5*||x/8||^2), per head.
// Block: 64 rows x 1 head, 128 threads = 4 waves (16 rows each).
// ---------------------------------------------------------------------------
__global__ __launch_bounds__(128) void feature_map_kernel(
    const float* __restrict__ X, const float* __restrict__ Rt,
    __bf16* __restrict__ XP, float* __restrict__ zsum, int computeZ) {
  __shared__ __align__(16) __bf16 sQ[64 * 72];
  __shared__ __align__(16) __bf16 sR[64 * 72];
  __shared__ float sSq[128];
  __shared__ float sZ[64];
  const int t = threadIdx.x;
  const int lane = t & 31;
  const int w = t >> 5;
  const int m0 = blockIdx.x * 64;
  const int h = blockIdx.y;

  {  // stage scaled X tile + partial row norms
    int row = t >> 1, half = t & 1;
    const float* src = X + (size_t)(m0 + row) * DIMV + h * HD + half * 32;
    float sq = 0.f;
#pragma unroll
    for (int j = 0; j < 8; ++j) {
      float s4[4];
      load4(src + j * 4, s4);
#pragma unroll
      for (int i = 0; i < 4; ++i) s4[i] *= 0.125f;
      sq += s4[0] * s4[0] + s4[1] * s4[1] + s4[2] * s4[2] + s4[3] * s4[3];
      store4bf(&sQ[row * 72 + half * 32 + j * 4], s4);
    }
    sSq[t] = sq;
  }
#pragma unroll
  for (int it = 0; it < 8; ++it) {  // stage Rt (Bt layout), vectorized
    int idx = it * 128 + t;
    int r = idx >> 4, c4 = (idx & 15) * 4;
    float s4[4];
    load4(Rt + r * 64 + c4, s4);
    store4bf(&sR[r * 72 + c4], s4);
  }
  if (computeZ && t < 64) sZ[t] = 0.f;
  __syncthreads();
  if (t < 64) sSq[t] = sSq[2 * t] + sSq[2 * t + 1];
  __syncthreads();

  v8f acc[4];
#pragma unroll
  for (int j = 0; j < 4; ++j) acc[j] = zero8();
#pragma unroll
  for (int kk = 0; kk < 2; ++kk) {
    v16bf a = frag_a(sQ, 72, w * 16, kk * 32, lane);
#pragma unroll
    for (int nt = 0; nt < 4; ++nt) {
      v16bf b = frag_b(sR, 72, nt * 16, kk * 32, lane);
      acc[nt] = bwmma(a, b, acc[nt]);
    }
  }
  const int hi8 = (lane >= 16) ? 8 : 0;
  const int nl = lane & 15;
#pragma unroll
  for (int nt = 0; nt < 4; ++nt)
#pragma unroll
    for (int r = 0; r < 8; ++r) {
      int rl = w * 16 + hi8 + r;
      float val = expf(acc[nt][r] - 0.5f * sSq[rl]);
      XP[(size_t)(m0 + rl) * DIMV + h * HD + nt * 16 + nl] = (__bf16)val;
      if (computeZ) atomicAdd(&sZ[rl], val);
    }
  if (computeZ) {
    __syncthreads();
    if (t < 64) zsum[(size_t)(m0 + t) * HEADS + h] = sZ[t];
  }
}

// ---------------------------------------------------------------------------
// kvt[bh][e][d] += sum_n v[n][e] * k'[n][d]   (split-K over sequence)
// Block: 128 n-chunk per (b,h), 128 threads = 4 waves (16 e-rows each).
// ---------------------------------------------------------------------------
__global__ __launch_bounds__(128) void kv_reduce_kernel(
    const __bf16* __restrict__ KP, const float* __restrict__ V,
    float* __restrict__ kvt) {
  __shared__ __align__(16) __bf16 sV[64 * 136];  // [e][n]  (v^T)
  __shared__ __align__(16) __bf16 sK[64 * 136];  // [d][n]  (k'^T)
  const int t = threadIdx.x;
  const int lane = t & 31;
  const int w = t >> 5;
  const int n0 = blockIdx.x * 128;
  const int bh = blockIdx.y;
  const int b = bh >> 4, h = bh & 15;
  const size_t rowbase = (size_t)b * SEQL + n0;

#pragma unroll
  for (int it = 0; it < 16; ++it) {  // vector reads, transposed LDS scatter
    int idx = it * 128 + t;
    int n = idx >> 4, e4 = (idx & 15) * 4;
    float v4[4], k4[4];
    load4(V + (rowbase + n) * DIMV + h * HD + e4, v4);
    load4(KP + (rowbase + n) * DIMV + h * HD + e4, k4);
#pragma unroll
    for (int i = 0; i < 4; ++i) {
      sV[(e4 + i) * 136 + n] = (__bf16)v4[i];
      sK[(e4 + i) * 136 + n] = (__bf16)k4[i];
    }
  }
  __syncthreads();

  v8f acc[4];
#pragma unroll
  for (int j = 0; j < 4; ++j) acc[j] = zero8();
#pragma unroll
  for (int kk = 0; kk < 4; ++kk) {
    v16bf a = frag_a(sV, 136, w * 16, kk * 32, lane);
#pragma unroll
    for (int nt = 0; nt < 4; ++nt) {
      v16bf b2 = frag_b(sK, 136, nt * 16, kk * 32, lane);
      acc[nt] = bwmma(a, b2, acc[nt]);
    }
  }
  const int hi8 = (lane >= 16) ? 8 : 0;
  const int nl = lane & 15;
#pragma unroll
  for (int nt = 0; nt < 4; ++nt)
#pragma unroll
    for (int r = 0; r < 8; ++r) {
      int e = w * 16 + hi8 + r, d = nt * 16 + nl;
      atomicAdd(&kvt[((size_t)bh * 64 + e) * 64 + d], acc[nt][r]);
    }
}

// ---------------------------------------------------------------------------
// out[n][h*64+e] = (1/(z+eps)) * sum_d q'[n][h*64+d] * kv[d][e]
// kvt[e][d] == kv[d][e] so kvt rows are exactly the Bt layout.
// ---------------------------------------------------------------------------
__global__ __launch_bounds__(128) void attn_out_kernel(
    const __bf16* __restrict__ QP, const float* __restrict__ kvt,
    const float* __restrict__ zsum, __bf16* __restrict__ OUT) {
  __shared__ __align__(16) __bf16 sQ[64 * 72];
  __shared__ __align__(16) __bf16 sKV[64 * 72];  // [e][d]
  const int t = threadIdx.x;
  const int lane = t & 31;
  const int w = t >> 5;
  const int m0 = blockIdx.x * 64;
  const int h = blockIdx.y;
  const int b = m0 >> 12;  // m0 / SEQL

#pragma unroll
  for (int it = 0; it < 8; ++it) {
    int idx = it * 128 + t;
    int r = idx >> 4, c4 = (idx & 15) * 4;
    // q' tile: raw 8-byte bf16 copy, no conversion
    *(uint2*)&sQ[r * 72 + c4] =
        *(const uint2*)&QP[(size_t)(m0 + r) * DIMV + h * HD + c4];
    float s4[4];
    load4(kvt + ((size_t)(b * HEADS + h) * 64 + r) * 64 + c4, s4);
    store4bf(&sKV[r * 72 + c4], s4);
  }
  __syncthreads();

  v8f acc[4];
#pragma unroll
  for (int j = 0; j < 4; ++j) acc[j] = zero8();
#pragma unroll
  for (int kk = 0; kk < 2; ++kk) {
    v16bf a = frag_a(sQ, 72, w * 16, kk * 32, lane);
#pragma unroll
    for (int nt = 0; nt < 4; ++nt) {
      v16bf b2 = frag_b(sKV, 72, nt * 16, kk * 32, lane);
      acc[nt] = bwmma(a, b2, acc[nt]);
    }
  }
  const int hi8 = (lane >= 16) ? 8 : 0;
  const int nl = lane & 15;
#pragma unroll
  for (int nt = 0; nt < 4; ++nt)
#pragma unroll
    for (int r = 0; r < 8; ++r) {
      int rl = w * 16 + hi8 + r;
      float zi = 1.0f / (zsum[(size_t)(m0 + rl) * HEADS + h] + 1e-8f);
      OUT[(size_t)(m0 + rl) * DIMV + h * HD + nt * 16 + nl] =
          (__bf16)(acc[nt][r] * zi);
    }
}

__global__ void transpose_R(const float* __restrict__ R, float* __restrict__ Rt) {
  int t = threadIdx.x;
#pragma unroll
  for (int it = 0; it < 16; ++it) {
    int idx = it * 256 + t;
    int r = idx >> 6, c = idx & 63;
    Rt[c * 64 + r] = R[r * 64 + c];
  }
}

__global__ void zero_f32(float* __restrict__ p, int n) {
  int i = blockIdx.x * blockDim.x + threadIdx.x;
  if (i < n) p[i] = 0.f;
}

extern "C" void kernel_launch(void* const* d_in, const int* in_sizes, int n_in,
                              void* d_out, int out_size, void* d_ws,
                              size_t ws_size, hipStream_t stream) {
  const float* x = (const float*)d_in[0];
  const float* Wq = (const float*)d_in[1];
  const float* bq = (const float*)d_in[2];
  const float* Wk = (const float*)d_in[3];
  const float* bk = (const float*)d_in[4];
  const float* Wv = (const float*)d_in[5];
  const float* bv = (const float*)d_in[6];
  const float* Wo = (const float*)d_in[7];
  const float* bo = (const float*)d_in[8];
  const float* R = (const float*)d_in[9];
  float* out = (float*)d_out;

  char* w = (char*)d_ws;
  size_t off = 0;
  float* Q = (float*)(w + off);    off += (size_t)M_TOT * DIMV * 4;
  float* Kf = (float*)(w + off);   off += (size_t)M_TOT * DIMV * 4;
  float* V = (float*)(w + off);    off += (size_t)M_TOT * DIMV * 4;
  __bf16* qp = (__bf16*)(w + off); off += (size_t)M_TOT * DIMV * 2;
  __bf16* kp = (__bf16*)(w + off); off += (size_t)M_TOT * DIMV * 2;
  __bf16* at = (__bf16*)(w + off); off += (size_t)M_TOT * DIMV * 2;
  float* kvt = (float*)(w + off);  off += (size_t)BATCH * HEADS * HD * HD * 4;
  float* zs = (float*)(w + off);   off += (size_t)M_TOT * HEADS * 4;
  float* Rt = (float*)(w + off);   off += 64 * 64 * 4;
  (void)off; (void)in_sizes; (void)n_in; (void)out_size; (void)ws_size;

  const int NKV = BATCH * HEADS * HD * HD;
  transpose_R<<<1, 256, 0, stream>>>(R, Rt);
  zero_f32<<<(NKV + 255) / 256, 256, 0, stream>>>(kvt, NKV);

  dim3 gg(DIMV / 128, M_TOT / 256);
  gemm_xwt_bias<float><<<gg, 256, 0, stream>>>(x, Wq, bq, Q, M_TOT, DIMV, DIMV);
  gemm_xwt_bias<float><<<gg, 256, 0, stream>>>(x, Wk, bk, Kf, M_TOT, DIMV, DIMV);
  gemm_xwt_bias<float><<<gg, 256, 0, stream>>>(x, Wv, bv, V, M_TOT, DIMV, DIMV);

  dim3 gf(M_TOT / 64, HEADS);
  feature_map_kernel<<<gf, 128, 0, stream>>>(Q, Rt, qp, zs, 1);
  feature_map_kernel<<<gf, 128, 0, stream>>>(Kf, Rt, kp, zs, 0);

  dim3 gk(SEQL / 128, BATCH * HEADS);
  kv_reduce_kernel<<<gk, 128, 0, stream>>>(kp, V, kvt);

  attn_out_kernel<<<gf, 128, 0, stream>>>(qp, kvt, zs, at);

  gemm_xwt_bias<__bf16><<<gg, 256, 0, stream>>>(at, Wo, bo, out, M_TOT, DIMV,
                                                DIMV);
}
